// FeatureEnhancement_74010876444799
// MI455X (gfx1250) — compile-verified, matrix operand order checked by reading, not run on hardware
//
#include <hip/hip_runtime.h>
#include <hip/hip_bf16.h>

// ---------------- problem dims ----------------
#define B_     8
#define CIN    1024
#define HW     9216      // 96*96
#define COUT   256
#define HID    64        // cin/16
#define NS     4         // NUM_SHARED
#define NGRP   32
#define CPG    8         // COUT/NGRP
#define EPSV   1e-5f

#define KSTEP  32
#define BM     64
#define BN     256

typedef __attribute__((ext_vector_type(16))) __bf16        bf16x16;
typedef __attribute__((ext_vector_type(8)))  float          f32x8;
typedef __attribute__((ext_vector_type(4)))  float          f32x4;
typedef __attribute__((ext_vector_type(8)))  unsigned short u16x8;
typedef __attribute__((ext_vector_type(2)))  unsigned int   u32x2;
typedef __attribute__((ext_vector_type(4)))  int            v4i;

union V16BF { bf16x16 v; u16x8 h[2]; };

#if __has_builtin(__builtin_amdgcn_global_load_async_to_lds_b128) && \
    __has_builtin(__builtin_amdgcn_s_wait_asynccnt)
#define HAS_ASYNC 1
#else
#define HAS_ASYNC 0
#endif

// pack two f32 -> bf16 pair in one dword (lo in [15:0], hi in [31:16]).
// round-half-up bias (+0x8000) then one v_perm_b32 picks both high halves.
__device__ __forceinline__ unsigned int pack_bf16(float lo, float hi) {
  union { float f; unsigned int u; } a, b;
  a.f = lo; b.f = hi;
  unsigned int ul = a.u + 0x8000u;
  unsigned int uh = b.u + 0x8000u;
  return __builtin_amdgcn_perm(uh, ul, 0x07060302u);
}

// ---------------- 1) global average pool ----------------
__global__ __launch_bounds__(256) void pool_kernel(const float* __restrict__ feat,
                                                   float* __restrict__ pooled) {
  __shared__ float red[256];
  int bc = blockIdx.x;                        // b*CIN + c
  const float* p = feat + (size_t)bc * HW;
  float s = 0.f;
  for (int i = threadIdx.x * 4; i < HW; i += 256 * 4) {
    f32x4 v = *(const f32x4*)(p + i);
    s += (v.x + v.y) + (v.z + v.w);
  }
  red[threadIdx.x] = s;
  __syncthreads();
  for (int off = 128; off > 0; off >>= 1) {
    if (threadIdx.x < off) red[threadIdx.x] += red[threadIdx.x + off];
    __syncthreads();
  }
  if (threadIdx.x == 0) pooled[bc] = red[0] * (1.0f / HW);
}

// ---------------- 2) SE MLP ----------------
__global__ __launch_bounds__(512) void se_kernel(const float* __restrict__ pooled,
                                                 const float* __restrict__ w1,
                                                 const float* __restrict__ b1,
                                                 const float* __restrict__ w2,
                                                 const float* __restrict__ b2,
                                                 float* __restrict__ svals) {
  __shared__ float hid[B_ * HID];
  int t = threadIdx.x;
  {
    int bb = t >> 6, j = t & 63;
    const float* pr = pooled + bb * CIN;
    const float* wr = w1 + j * CIN;
    float acc = b1[j];
    for (int i = 0; i < CIN; ++i) acc += pr[i] * wr[i];
    hid[bb * HID + j] = fmaxf(acc, 0.f);
  }
  __syncthreads();
  if (t < B_ * NS) {
    int bb = t >> 2, sh = t & 3;
    const float* hr = hid + bb * HID;
    const float* wr = w2 + sh * HID;
    float acc = b2[sh];
    for (int i = 0; i < HID; ++i) acc += hr[i] * wr[i];
    svals[bb * NS + sh] = 1.f / (1.f + __expf(-acc));
  }
}

// ---------------- 3) per-sample bf16 A = s[b, i/256] * w_red ----------------
__global__ __launch_bounds__(256) void scalew_kernel(const float* __restrict__ w_red,
                                                     const float* __restrict__ svals,
                                                     unsigned short* __restrict__ Abf) {
  int idx = (blockIdx.x * 256 + threadIdx.x) * 4;
  int b   = idx >> 18;
  int oi  = idx & ((COUT * CIN) - 1);
  int i   = oi & (CIN - 1);
  float s = svals[b * NS + (i >> 8)];
  f32x4 w = *(const f32x4*)(w_red + oi);
  u32x2 o;
  o.x = pack_bf16(w.x * s, w.y * s);
  o.y = pack_bf16(w.z * s, w.w * s);
  *(u32x2*)(Abf + idx) = o;
}

// ---------------- 4) bf16 WMMA GEMM, double-buffered pipeline ----------------
// block = 256 thr = 8 wave32 (2x4). block tile 64x256, wave tile 32x64, K-step 32.
__global__ __launch_bounds__(256) void gemm_kernel(const float* __restrict__ feat,
                                                   const unsigned short* __restrict__ Abf,
                                                   float* __restrict__ out) {
  __shared__ __align__(16) unsigned short lds_a[2][BM][32];   // 8 KB
  __shared__ __align__(16) unsigned short lds_b[2][BN][40];   // 40 KB, 80B rows (16B mult)

  const int nbase = blockIdx.x * BN;
  const int mbase = blockIdx.y * BM;
  const int b     = blockIdx.z;
  const float*          Bm = feat + (size_t)b * CIN * HW;
  const unsigned short* Am = Abf  + (size_t)b * COUT * CIN;

  const int t    = threadIdx.x;
  const int lane = t & 31;
  const int w    = t >> 5;
  const int wm   = w & 1;        // 0..1 -> 32-row M sub-tile
  const int wn   = w >> 1;       // 0..3 -> 64-col N sub-tile

  // A staging map: 64 rows x 32 k, one 16B chunk per thread
  const int ar  = t >> 2;
  const int akq = (t & 3) * 8;
  // B staging map: 2 k-rows x 16 n per thread
  const int kp = (t >> 4) * 2;        // 0,2,..,30
  const int nq = (t & 15) * 16;       // 0..240

  const int lr = lane & 15;
  const int lk = (lane >> 4) * 8;

  f32x8 c[2][4] = {};
  float v0[16], v1[16];

  // ---- prologue: A(0) -> lds_a[0] (async DMA), B(0) -> regs ----
#if HAS_ASYNC
  __builtin_amdgcn_global_load_async_to_lds_b128(
      (v4i*)(Am + (size_t)(mbase + ar) * CIN + akq),
      (__attribute__((address_space(3))) v4i*)&lds_a[0][ar][akq], 0, 0);
#else
  *(u16x8*)&lds_a[0][ar][akq] =
      *(const u16x8*)(Am + (size_t)(mbase + ar) * CIN + akq);
#endif
  {
    const float* r0 = Bm + (size_t)kp * HW + nbase + nq;
    const float* r1 = r0 + HW;
#pragma unroll
    for (int q = 0; q < 4; ++q) {
      f32x4 x0 = *(const f32x4*)(r0 + q * 4);
      f32x4 x1 = *(const f32x4*)(r1 + q * 4);
      v0[q*4+0]=x0.x; v0[q*4+1]=x0.y; v0[q*4+2]=x0.z; v0[q*4+3]=x0.w;
      v1[q*4+0]=x1.x; v1[q*4+1]=x1.y; v1[q*4+2]=x1.z; v1[q*4+3]=x1.w;
    }
  }

  int buf = 0;
  for (int kk = 0; kk < CIN; kk += KSTEP) {
    // ---- commit B(i) regs to lds_b[buf] (transposed [n][k], packed k-pairs) ----
#pragma unroll
    for (int j = 0; j < 16; ++j) {
      *(unsigned int*)&lds_b[buf][nq + j][kp] = pack_bf16(v0[j], v1[j]);
    }
#if HAS_ASYNC
    __builtin_amdgcn_s_wait_asynccnt(0);     // A(i) landed in lds_a[buf]
#endif
    __syncthreads();

    // ---- prefetch tile i+1 (overlaps WMMAs below) ----
    const int nk = kk + KSTEP;
    if (nk < CIN) {
#if HAS_ASYNC
      __builtin_amdgcn_global_load_async_to_lds_b128(
          (v4i*)(Am + (size_t)(mbase + ar) * CIN + nk + akq),
          (__attribute__((address_space(3))) v4i*)&lds_a[buf ^ 1][ar][akq], 0, 0);
#else
      *(u16x8*)&lds_a[buf ^ 1][ar][akq] =
          *(const u16x8*)(Am + (size_t)(mbase + ar) * CIN + nk + akq);
#endif
      const float* r0 = Bm + (size_t)(nk + kp) * HW + nbase + nq;
      const float* r1 = r0 + HW;
#pragma unroll
      for (int q = 0; q < 4; ++q) {
        f32x4 x0 = *(const f32x4*)(r0 + q * 4);
        f32x4 x1 = *(const f32x4*)(r1 + q * 4);
        v0[q*4+0]=x0.x; v0[q*4+1]=x0.y; v0[q*4+2]=x0.z; v0[q*4+3]=x0.w;
        v1[q*4+0]=x1.x; v1[q*4+1]=x1.y; v1[q*4+2]=x1.z; v1[q*4+3]=x1.w;
      }
      // speculative prefetch of tile i+2 feat rows (global_prefetch_b8)
      if (nk + KSTEP < CIN) {
        __builtin_prefetch(Bm + (size_t)(nk + KSTEP + kp) * HW + nbase + nq, 0, 1);
        __builtin_prefetch(Bm + (size_t)(nk + KSTEP + kp + 1) * HW + nbase + nq, 0, 1);
      }
    }

    // ---- load all 6 fragments, then 8 independent back-to-back WMMAs ----
    V16BF af0, af1, bf0, bf1, bf2, bf3;
    af0.h[0] = *(const u16x8*)&lds_a[buf][wm * 32 +      lr][lk];
    af0.h[1] = *(const u16x8*)&lds_a[buf][wm * 32 +      lr][lk + 16];
    af1.h[0] = *(const u16x8*)&lds_a[buf][wm * 32 + 16 + lr][lk];
    af1.h[1] = *(const u16x8*)&lds_a[buf][wm * 32 + 16 + lr][lk + 16];
    const int nb0 = wn * 64 + lr;
    bf0.h[0] = *(const u16x8*)&lds_b[buf][nb0     ][lk];
    bf0.h[1] = *(const u16x8*)&lds_b[buf][nb0     ][lk + 16];
    bf1.h[0] = *(const u16x8*)&lds_b[buf][nb0 + 16][lk];
    bf1.h[1] = *(const u16x8*)&lds_b[buf][nb0 + 16][lk + 16];
    bf2.h[0] = *(const u16x8*)&lds_b[buf][nb0 + 32][lk];
    bf2.h[1] = *(const u16x8*)&lds_b[buf][nb0 + 32][lk + 16];
    bf3.h[0] = *(const u16x8*)&lds_b[buf][nb0 + 48][lk];
    bf3.h[1] = *(const u16x8*)&lds_b[buf][nb0 + 48][lk + 16];

    c[0][0] = __builtin_amdgcn_wmma_f32_16x16x32_bf16(false, af0.v, false, bf0.v, (short)0, c[0][0], false, false);
    c[1][0] = __builtin_amdgcn_wmma_f32_16x16x32_bf16(false, af1.v, false, bf0.v, (short)0, c[1][0], false, false);
    c[0][1] = __builtin_amdgcn_wmma_f32_16x16x32_bf16(false, af0.v, false, bf1.v, (short)0, c[0][1], false, false);
    c[1][1] = __builtin_amdgcn_wmma_f32_16x16x32_bf16(false, af1.v, false, bf1.v, (short)0, c[1][1], false, false);
    c[0][2] = __builtin_amdgcn_wmma_f32_16x16x32_bf16(false, af0.v, false, bf2.v, (short)0, c[0][2], false, false);
    c[1][2] = __builtin_amdgcn_wmma_f32_16x16x32_bf16(false, af1.v, false, bf2.v, (short)0, c[1][2], false, false);
    c[0][3] = __builtin_amdgcn_wmma_f32_16x16x32_bf16(false, af0.v, false, bf3.v, (short)0, c[0][3], false, false);
    c[1][3] = __builtin_amdgcn_wmma_f32_16x16x32_bf16(false, af1.v, false, bf3.v, (short)0, c[1][3], false, false);

    buf ^= 1;
  }

  // ---- epilogue: C layout — VGPR j, lane l => M = j + 8*(l>>4), N = l&15 ----
#pragma unroll
  for (int fm = 0; fm < 2; ++fm) {
    const int rowbase = mbase + wm * 32 + fm * 16 + 8 * (lane >> 4);
#pragma unroll
    for (int f = 0; f < 4; ++f) {
      const int col = nbase + wn * 64 + f * 16 + lr;
      float* op = out + ((size_t)b * COUT + rowbase) * HW + col;
#pragma unroll
      for (int j = 0; j < 8; ++j) op[(size_t)j * HW] = c[fm][f][j];
    }
  }
}

// ---------------- 5) deterministic GN stats ----------------
__global__ __launch_bounds__(256) void gn_stats_kernel(const float* __restrict__ x,
                                                       float* __restrict__ st) {
  __shared__ float rs[256], rq[256];
  int bg = blockIdx.x;
  const float* p = x + (size_t)bg * CPG * HW;
  float s = 0.f, q = 0.f;
  for (int i = threadIdx.x * 4; i < CPG * HW; i += 256 * 4) {
    f32x4 v = *(const f32x4*)(p + i);
    s += (v.x + v.y) + (v.z + v.w);
    q += (v.x * v.x + v.y * v.y) + (v.z * v.z + v.w * v.w);
  }
  rs[threadIdx.x] = s; rq[threadIdx.x] = q;
  __syncthreads();
  for (int off = 128; off > 0; off >>= 1) {
    if (threadIdx.x < off) {
      rs[threadIdx.x] += rs[threadIdx.x + off];
      rq[threadIdx.x] += rq[threadIdx.x + off];
    }
    __syncthreads();
  }
  if (threadIdx.x == 0) {
    const float inv = 1.0f / (CPG * HW);
    float mu  = rs[0] * inv;
    float var = rq[0] * inv - mu * mu;
    st[bg * 2 + 0] = mu;
    st[bg * 2 + 1] = rsqrtf(var + EPSV);
  }
}

// ---------------- 6) GN apply + affine + relu ----------------
__global__ __launch_bounds__(256) void gn_apply_kernel(float* __restrict__ x,
                                                       const float* __restrict__ st,
                                                       const float* __restrict__ gamma,
                                                       const float* __restrict__ beta) {
  size_t idx = ((size_t)blockIdx.x * 256 + threadIdx.x) * 4;
  int c  = (int)((idx / HW) & (COUT - 1));
  int b  = (int)(idx / ((size_t)HW * COUT));
  int bg = b * NGRP + (c >> 3);
  float mu = st[bg * 2 + 0], rsq = st[bg * 2 + 1];
  float g  = gamma[c] * rsq;
  float be = beta[c] - mu * g;
  f32x4 v = *(f32x4*)(x + idx);
  v.x = fmaxf(v.x * g + be, 0.f);
  v.y = fmaxf(v.y * g + be, 0.f);
  v.z = fmaxf(v.z * g + be, 0.f);
  v.w = fmaxf(v.w * g + be, 0.f);
  *(f32x4*)(x + idx) = v;
}

extern "C" void kernel_launch(void* const* d_in, const int* in_sizes, int n_in,
                              void* d_out, int out_size, void* d_ws, size_t ws_size,
                              hipStream_t stream) {
  const float* feat  = (const float*)d_in[0];
  const float* w1    = (const float*)d_in[1];
  const float* b1    = (const float*)d_in[2];
  const float* w2    = (const float*)d_in[3];
  const float* b2    = (const float*)d_in[4];
  const float* w_red = (const float*)d_in[5];
  const float* gamma = (const float*)d_in[6];
  const float* beta  = (const float*)d_in[7];
  float* out = (float*)d_out;

  char* ws = (char*)d_ws;
  unsigned short* Abf = (unsigned short*)ws;                 // 4 MB
  float* pooled = (float*)(ws + 4194304);                    // 8192 f32
  float* svals  = (float*)(ws + 4194304 + 32768);            // 32 f32
  float* st     = (float*)(ws + 4194304 + 32768 + 128);      // 512 f32

  pool_kernel  <<<B_ * CIN, 256, 0, stream>>>(feat, pooled);
  se_kernel    <<<1, 512, 0, stream>>>(pooled, w1, b1, w2, b2, svals);
  scalew_kernel<<<(B_ * COUT * CIN) / (256 * 4), 256, 0, stream>>>(w_red, svals, Abf);
  gemm_kernel  <<<dim3(HW / BN, COUT / BM, B_), 256, 0, stream>>>(feat, Abf, out);
  gn_stats_kernel<<<B_ * NGRP, 256, 0, stream>>>(out, st);
  gn_apply_kernel<<<(B_ * COUT * HW) / (256 * 4), 256, 0, stream>>>(out, st, gamma, beta);
}
